// GlobalNodeAttention_50208167690484
// MI455X (gfx1250) — compile-verified, matrix operand order checked by reading, not run on hardware
//
#include <hip/hip_runtime.h>
#include <stdint.h>

// Problem constants (match reference)
#define D_MODEL   256
#define NUM_HEADS 8
#define HEAD_DIM  32
#define ROPE_DIM  16
#define B_GRAPHS  16
#define NMAX      512
#define BN        (B_GRAPHS * NMAX)   // 8192 padded rows

typedef __attribute__((ext_vector_type(16))) __bf16 v16bf;
typedef __attribute__((ext_vector_type(8)))  float  v8f;

union Frag16 { v16bf v; uint32_t u[8]; };

__device__ __forceinline__ unsigned short f2bf(float f) {
  union { float f; uint32_t u; } c; c.f = f;
  uint32_t u = c.u;
  u += 0x7FFFu + ((u >> 16) & 1u);   // round-to-nearest-even
  return (unsigned short)(u >> 16);
}
__device__ __forceinline__ uint32_t pack2bf(float lo, float hi) {
  return (uint32_t)f2bf(lo) | ((uint32_t)f2bf(hi) << 16);
}
// 16-bit A-matrix (16xK) fragment K-base for VGPR v, lane-half `half`
// (ISA 7.12.2: V0..3 hold K=0..7|8..15, V4..7 hold K=16..23|24..31)
__device__ __forceinline__ int kbase16(int v, int half) {
  return (v < 4 ? 2 * v : 16 + 2 * (v - 4)) + 8 * half;
}

// ---------------- K0: ragged offsets (batch is sorted) ----------------
__global__ void k_offsets(const int* __restrict__ batch, int N, int* __restrict__ off) {
  for (int i = threadIdx.x; i < N; i += blockDim.x) {
    if (i == 0 || batch[i] != batch[i - 1]) off[batch[i]] = i;
    if (i == N - 1) off[B_GRAPHS] = N;
  }
}

// ---------------- K1: pad x (bf16) and pos (+valid flag) ----------------
__global__ void k_pad(const float* __restrict__ x, const float* __restrict__ pos,
                      const int* __restrict__ off,
                      unsigned short* __restrict__ xh, float* __restrict__ pos_pad) {
  int row = blockIdx.x;                 // 0..BN-1
  int b = row >> 9, p = row & (NMAX - 1);
  int o0 = off[b], o1 = off[b + 1];
  bool valid = p < (o1 - o0);
  int src = o0 + p;
  int c = threadIdx.x;
  xh[row * D_MODEL + c] = valid ? f2bf(x[(size_t)src * D_MODEL + c]) : (unsigned short)0;
  if (c == 0) {
    float px = 0.f, py = 0.f, pz = 0.f, fl = 0.f;
    if (valid) { px = pos[src * 3]; py = pos[src * 3 + 1]; pz = pos[src * 3 + 2]; fl = 1.f; }
    pos_pad[row * 4 + 0] = px; pos_pad[row * 4 + 1] = py;
    pos_pad[row * 4 + 2] = pz; pos_pad[row * 4 + 3] = fl;
  }
}

// ---------------- K1b: transpose weights to bf16 (K contiguous per column) ----------------
__global__ void k_wconv(const float* __restrict__ Wqkv, const float* __restrict__ Wout,
                        unsigned short* __restrict__ WqkvT, unsigned short* __restrict__ WoutT) {
  int idx = blockIdx.x * blockDim.x + threadIdx.x;
  const int n1 = 768 * 256;
  if (idx < n1) {
    int n = idx >> 8, k = idx & 255;
    WqkvT[n * 256 + k] = f2bf(Wqkv[k * 768 + n]);
  } else {
    int j = idx - n1;
    if (j < 256 * 256) {
      int n = j >> 8, k = j & 255;
      WoutT[n * 256 + k] = f2bf(Wout[k * 256 + n]);
    }
  }
}

// ---------------- K2: QKV GEMM (WMMA bf16), scatter to head-major Q/K and transposed V ----
__global__ void k_qkv_gemm(const unsigned short* __restrict__ xh,
                           const unsigned short* __restrict__ WqkvT,
                           unsigned short* __restrict__ Qh,
                           unsigned short* __restrict__ Kh,
                           unsigned short* __restrict__ Vt) {
  int lane = threadIdx.x & 31;
  int wave = threadIdx.x >> 5;
  int tile = blockIdx.x * 4 + wave;     // (BN/16) * (768/16) = 24576 tiles
  int mt = tile / 48, nt = tile - mt * 48;
  int m0 = mt * 16, n0 = nt * 16;
  int half = lane >> 4, ln = lane & 15;
  int arow = m0 + ln;
  int bcol = n0 + ln;
  v8f acc = {};
  for (int k0 = 0; k0 < 256; k0 += 32) {
    Frag16 a, bf;
#pragma unroll
    for (int v = 0; v < 8; v++) {
      int kb = k0 + kbase16(v, half);
      a.u[v]  = *reinterpret_cast<const uint32_t*>(&xh[(size_t)arow * 256 + kb]);
      bf.u[v] = *reinterpret_cast<const uint32_t*>(&WqkvT[(size_t)bcol * 256 + kb]);
    }
    acc = __builtin_amdgcn_wmma_f32_16x16x32_bf16(false, a.v, false, bf.v,
                                                  (short)0, acc, false, false);
  }
#pragma unroll
  for (int p = 0; p < 8; p++) {
    int row = m0 + p + 8 * half;        // padded node row
    int c = n0 + ln;                    // 0..767
    int part = c >> 8;                  // 0=q 1=k 2=v
    int cc = c & 255;
    int h = cc >> 5, dd = cc & 31;
    int bb = row >> 9, r = row & (NMAX - 1);
    unsigned short val = f2bf(acc[p]);
    int bh = bb * NUM_HEADS + h;
    if (part == 0)      Qh[((size_t)bh * NMAX + r) * HEAD_DIM + dd] = val;
    else if (part == 1) Kh[((size_t)bh * NMAX + r) * HEAD_DIM + dd] = val;
    else                Vt[((size_t)bh * HEAD_DIM + dd) * NMAX + r] = val;
  }
}

// ---------------- K3: fused attention (flash-style, one wave per 16-query tile) ------------
__global__ void k_attn(const unsigned short* __restrict__ Qh,
                       const unsigned short* __restrict__ Kh,
                       const unsigned short* __restrict__ Vt,
                       const float* __restrict__ pos_pad,
                       const float* __restrict__ rope_freqs,
                       const float* __restrict__ W_rope,
                       const int* __restrict__ off,
                       float* __restrict__ attn_out) {
  __shared__ unsigned short Plds[4][16 * 32];   // per-wave P staging (C-layout -> A-layout)
  int lane = threadIdx.x & 31;
  int wave = threadIdx.x >> 5;
  int tile = blockIdx.x * 4 + wave;             // B*H*(NMAX/16) = 4096 tiles
  int itile = tile & 31;
  int bh = tile >> 5;
  int h = bh & 7, b = bh >> 3;
  int i0 = itile * 16;
  int half = lane >> 4, ln = lane & 15;
  const float scale = 0.17677669529663687f;     // 1/sqrt(32)

  float af[ROPE_DIM], wr[ROPE_DIM];
#pragma unroll
  for (int r = 0; r < ROPE_DIM; r++) {
    af[r] = fabsf(rope_freqs[r]);
    wr[r] = W_rope[r * NUM_HEADS + h];
  }
  // query positions for this lane's C-fragment rows (M = p + 8*half)
  float qx[8], qy[8], qz[8];
#pragma unroll
  for (int p = 0; p < 8; p++) {
    int row = b * NMAX + i0 + p + 8 * half;
    qx[p] = pos_pad[row * 4 + 0];
    qy[p] = pos_pad[row * 4 + 1];
    qz[p] = pos_pad[row * 4 + 2];
  }
  // Q fragment (A-layout, hd == 32 == K exactly)
  Frag16 qa;
#pragma unroll
  for (int v = 0; v < 8; v++) {
    int kb = kbase16(v, half);
    qa.u[v] = *reinterpret_cast<const uint32_t*>(
        &Qh[((size_t)bh * NMAX + i0 + ln) * HEAD_DIM + kb]);
  }
  v8f o0 = {}, o1 = {};
  float rmax[8], rsum[8];
#pragma unroll
  for (int p = 0; p < 8; p++) { rmax[p] = -1e9f; rsum[p] = 0.f; }

  int nkeys = off[b + 1] - off[b];
  int njb = (nkeys + 31) >> 5;                  // uniform per wave
  for (int jb = 0; jb < njb; jb++) {
    int j0 = jb * 32;
    // K^T B-fragments for two 16-key columns
    Frag16 kb0, kb1;
#pragma unroll
    for (int v = 0; v < 8; v++) {
      int kk = kbase16(v, half);
      kb0.u[v] = *reinterpret_cast<const uint32_t*>(
          &Kh[((size_t)bh * NMAX + j0 + ln) * HEAD_DIM + kk]);
      kb1.u[v] = *reinterpret_cast<const uint32_t*>(
          &Kh[((size_t)bh * NMAX + j0 + 16 + ln) * HEAD_DIM + kk]);
    }
    v8f zero = {};
    v8f s0 = __builtin_amdgcn_wmma_f32_16x16x32_bf16(false, qa.v, false, kb0.v,
                                                     (short)0, zero, false, false);
    v8f s1 = __builtin_amdgcn_wmma_f32_16x16x32_bf16(false, qa.v, false, kb1.v,
                                                     (short)0, zero, false, false);
    // key positions / validity for this lane's columns
    int jc0 = (b * NMAX + j0 + ln) * 4;
    int jc1 = (b * NMAX + j0 + 16 + ln) * 4;
    float k0x = pos_pad[jc0 + 0], k0y = pos_pad[jc0 + 1], k0z = pos_pad[jc0 + 2], k0f = pos_pad[jc0 + 3];
    float k1x = pos_pad[jc1 + 0], k1y = pos_pad[jc1 + 1], k1z = pos_pad[jc1 + 2], k1f = pos_pad[jc1 + 3];

    float p0[8], p1[8];
#pragma unroll
    for (int p = 0; p < 8; p++) {
      // RoPE distance bias (16 cosines each), co-executes with XDL WMMA
      float dx = qx[p] - k0x, dy = qy[p] - k0y, dz = qz[p] - k0z;
      float d = sqrtf(dx * dx + dy * dy + dz * dz);
      float bias = 0.f;
#pragma unroll
      for (int r = 0; r < ROPE_DIM; r++) bias += __cosf(d * af[r]) * wr[r];
      float l0 = (k0f > 0.f) ? (s0[p] * scale + bias) : -1e9f;

      dx = qx[p] - k1x; dy = qy[p] - k1y; dz = qz[p] - k1z;
      d = sqrtf(dx * dx + dy * dy + dz * dz);
      bias = 0.f;
#pragma unroll
      for (int r = 0; r < ROPE_DIM; r++) bias += __cosf(d * af[r]) * wr[r];
      float l1 = (k1f > 0.f) ? (s1[p] * scale + bias) : -1e9f;

      // row reductions across the 16-lane half (C rows live across lanes)
      float m = fmaxf(l0, l1);
      m = fmaxf(m, __shfl_xor(m, 1));
      m = fmaxf(m, __shfl_xor(m, 2));
      m = fmaxf(m, __shfl_xor(m, 4));
      m = fmaxf(m, __shfl_xor(m, 8));
      float nm = fmaxf(rmax[p], m);
      float alpha = __expf(rmax[p] - nm);
      float e0 = __expf(l0 - nm);
      float e1 = __expf(l1 - nm);
      float rs = e0 + e1;
      rs += __shfl_xor(rs, 1);
      rs += __shfl_xor(rs, 2);
      rs += __shfl_xor(rs, 4);
      rs += __shfl_xor(rs, 8);
      rsum[p] = rsum[p] * alpha + rs;
      rmax[p] = nm;
      o0[p] *= alpha;
      o1[p] *= alpha;
      p0[p] = e0; p1[p] = e1;
    }
    // stage P (16x32 bf16) through per-wave LDS: C-layout -> A-layout
#pragma unroll
    for (int p = 0; p < 8; p++) {
      int m = p + 8 * half;
      Plds[wave][m * 32 + ln]      = f2bf(p0[p]);
      Plds[wave][m * 32 + 16 + ln] = f2bf(p1[p]);
    }
    __builtin_amdgcn_wave_barrier();
    asm volatile("s_wait_dscnt 0" ::: "memory");
    Frag16 pa;
#pragma unroll
    for (int v = 0; v < 8; v++) {
      int kk = kbase16(v, half);
      pa.u[v] = *reinterpret_cast<const uint32_t*>(&Plds[wave][ln * 32 + kk]);
    }
    // V B-fragments: columns = hd dims, rows = keys (V stored transposed -> contiguous pairs)
    Frag16 vb0, vb1;
#pragma unroll
    for (int v = 0; v < 8; v++) {
      int kk = kbase16(v, half);
      vb0.u[v] = *reinterpret_cast<const uint32_t*>(
          &Vt[((size_t)bh * HEAD_DIM + ln) * NMAX + j0 + kk]);
      vb1.u[v] = *reinterpret_cast<const uint32_t*>(
          &Vt[((size_t)bh * HEAD_DIM + 16 + ln) * NMAX + j0 + kk]);
    }
    o0 = __builtin_amdgcn_wmma_f32_16x16x32_bf16(false, pa.v, false, vb0.v,
                                                 (short)0, o0, false, false);
    o1 = __builtin_amdgcn_wmma_f32_16x16x32_bf16(false, pa.v, false, vb1.v,
                                                 (short)0, o1, false, false);
  }
#pragma unroll
  for (int p = 0; p < 8; p++) {
    float inv = (rsum[p] > 0.f) ? 1.f / rsum[p] : 0.f;
    int row = i0 + p + 8 * half;
    size_t base = ((size_t)b * NMAX + row) * D_MODEL + h * HEAD_DIM;
    attn_out[base + ln]      = o0[p] * inv;
    attn_out[base + 16 + ln] = o1[p] * inv;
  }
}

// ---------------- K4: output projection GEMM (WMMA bf16) ----------------
__global__ void k_out_gemm(const float* __restrict__ A,
                           const unsigned short* __restrict__ WoutT,
                           float* __restrict__ C) {
  int lane = threadIdx.x & 31, wave = threadIdx.x >> 5;
  int tile = blockIdx.x * 4 + wave;     // (BN/16)*(256/16) = 8192 tiles
  int mt = tile >> 4, nt = tile & 15;
  int m0 = mt * 16, n0 = nt * 16;
  int half = lane >> 4, ln = lane & 15;
  v8f acc = {};
  for (int k0 = 0; k0 < 256; k0 += 32) {
    Frag16 a, bf;
#pragma unroll
    for (int v = 0; v < 8; v++) {
      int kb = k0 + kbase16(v, half);
      a.u[v]  = pack2bf(A[(size_t)(m0 + ln) * 256 + kb],
                        A[(size_t)(m0 + ln) * 256 + kb + 1]);
      bf.u[v] = *reinterpret_cast<const uint32_t*>(&WoutT[(size_t)(n0 + ln) * 256 + kb]);
    }
    acc = __builtin_amdgcn_wmma_f32_16x16x32_bf16(false, a.v, false, bf.v,
                                                  (short)0, acc, false, false);
  }
#pragma unroll
  for (int p = 0; p < 8; p++)
    C[(size_t)(m0 + p + 8 * half) * 256 + n0 + ln] = acc[p];
}

// ---------------- K5: gather + residual + LayerNorm ----------------
__global__ void k_ln(const float* __restrict__ x, const float* __restrict__ o_pad,
                     const int* __restrict__ batch, const int* __restrict__ off,
                     const float* __restrict__ gamma, const float* __restrict__ beta,
                     float* __restrict__ out) {
  __shared__ float sbuf[256];
  int n = blockIdx.x, c = threadIdx.x;
  int b = batch[n];
  int p = n - off[b];
  float y = x[(size_t)n * 256 + c] + o_pad[((size_t)b * NMAX + p) * 256 + c];
  sbuf[c] = y; __syncthreads();
  for (int s = 128; s > 0; s >>= 1) { if (c < s) sbuf[c] += sbuf[c + s]; __syncthreads(); }
  float mu = sbuf[0] * (1.f / 256.f); __syncthreads();
  float d = y - mu;
  sbuf[c] = d * d; __syncthreads();
  for (int s = 128; s > 0; s >>= 1) { if (c < s) sbuf[c] += sbuf[c + s]; __syncthreads(); }
  float var = sbuf[0] * (1.f / 256.f);
  out[(size_t)n * 256 + c] = gamma[c] * d * rsqrtf(var + 1e-5f) + beta[c];
}

extern "C" void kernel_launch(void* const* d_in, const int* in_sizes, int n_in,
                              void* d_out, int out_size, void* d_ws, size_t ws_size,
                              hipStream_t stream) {
  (void)n_in; (void)out_size; (void)ws_size;
  const float* x     = (const float*)d_in[0];
  const float* pos   = (const float*)d_in[1];
  const int*   batch = (const int*)d_in[2];
  const float* Wqkv  = (const float*)d_in[3];
  const float* Wout  = (const float*)d_in[4];
  const float* rope  = (const float*)d_in[5];
  const float* Wr    = (const float*)d_in[6];
  const float* gamma = (const float*)d_in[7];
  const float* beta  = (const float*)d_in[8];
  int N = in_sizes[0] / D_MODEL;

  char* w = (char*)d_ws;
  auto carve = [&](size_t bytes) { char* r = w; w += (bytes + 255) & ~(size_t)255; return r; };
  int*            off      = (int*)carve((B_GRAPHS + 1) * sizeof(int));
  float*          pos_pad  = (float*)carve((size_t)BN * 4 * sizeof(float));
  unsigned short* WqkvT    = (unsigned short*)carve((size_t)768 * 256 * 2);
  unsigned short* WoutT    = (unsigned short*)carve((size_t)256 * 256 * 2);
  unsigned short* xh       = (unsigned short*)carve((size_t)BN * 256 * 2);
  unsigned short* Qh       = (unsigned short*)carve((size_t)BN * 256 * 2);
  unsigned short* Kh       = (unsigned short*)carve((size_t)BN * 256 * 2);
  unsigned short* Vt       = (unsigned short*)carve((size_t)BN * 256 * 2);
  float*          attn_out = (float*)carve((size_t)BN * 256 * 4);
  float*          o_pad    = (float*)carve((size_t)BN * 256 * 4);

  k_offsets<<<1, 256, 0, stream>>>(batch, N, off);
  k_pad<<<BN, 256, 0, stream>>>(x, pos, off, xh, pos_pad);
  k_wconv<<<1024, 256, 0, stream>>>(Wqkv, Wout, WqkvT, WoutT);
  k_qkv_gemm<<<6144, 128, 0, stream>>>(xh, WqkvT, Qh, Kh, Vt);
  k_attn<<<1024, 128, 0, stream>>>(Qh, Kh, Vt, pos_pad, rope, Wr, off, attn_out);
  k_out_gemm<<<2048, 128, 0, stream>>>(attn_out, WoutT, o_pad);
  k_ln<<<N, 256, 0, stream>>>(x, o_pad, batch, off, gamma, beta, (float*)d_out);
}
